// TransformerEncoderLayerWithChannels_47442208752131
// MI455X (gfx1250) — matrix operand order repaired and checked by reading, not compiled, not run on hardware
//
#include <hip/hip_runtime.h>

// ---------------------------------------------------------------------------
// TransformerEncoderLayerWithChannels for MI455X (gfx1250, wave32, WMMA).
// B=8 C=4 N=2048 E=512 F=2048. Compute-bound (~550 GFLOP dense GEMM vs ~12us
// of HBM traffic at 23.3 TB/s) -> bf16 operands + v_wmma_f32_16x16x32_bf16
// with fp32 accumulation.  K tiles are staged into LDS by the Tensor Data
// Mover (TENSOR_LOAD_TO_LDS + s_wait_tensorcnt), with DMA-inserted row
// padding to avoid LDS bank conflicts.
//
// Workspace layout (requires >= 128 MB):
//   xb  : bf16 copy of src            (64 MB)
//   s2b : bf16 LayerNorm1 output      (64 MB)
// ---------------------------------------------------------------------------

typedef __bf16 bf16_t;
typedef __attribute__((ext_vector_type(16))) __bf16 bf16x16;
typedef __attribute__((ext_vector_type(8)))  __bf16 bf16x8;
typedef __attribute__((ext_vector_type(4)))  __bf16 bf16x4;
typedef __attribute__((ext_vector_type(8)))  float  v8f;
typedef __attribute__((ext_vector_type(4)))  unsigned int u32x4;
typedef __attribute__((ext_vector_type(8)))  int i32x8;
typedef __attribute__((ext_vector_type(4)))  int i32x4;

static constexpr int Bd = 8, Cd = 4, Nd = 2048, Ed = 512, Fd = 2048;
static constexpr float EPSv = 1e-5f;
static constexpr float MASK_SCALE = 5.0f;

// ---- helpers ---------------------------------------------------------------

__device__ __forceinline__ v8f wmma_bf16(bf16x16 a, bf16x16 b, v8f c) {
  // v_wmma_f32_16x16x32_bf16  D = A(16x32) * B(32x16) + C
  return __builtin_amdgcn_wmma_f32_16x16x32_bf16(
      /*neg_a=*/false, a, /*neg_b=*/false, b,
      /*c_mod=*/(short)0, c, /*reuse_a=*/false, /*reuse_b=*/false);
}

// A-operand (16x32, 16-bit) per-lane fragment: row = lane&15,
// elements 0..7 <- k = klo+0..7, elements 8..15 <- k = klo+8..15,
// klo = (lane>=16)*16.  Caller passes the two 16B-aligned half pointers.
__device__ __forceinline__ bf16x16 ldA(const bf16_t* p_lo, const bf16_t* p_hi) {
  bf16x8 lo = *(const bf16x8*)p_lo;
  bf16x8 hi = *(const bf16x8*)p_hi;
  return __builtin_shufflevector(lo, hi, 0, 1, 2, 3, 4, 5, 6, 7,
                                 8, 9, 10, 11, 12, 13, 14, 15);
}

__device__ __forceinline__ float redmax16(float v) {
  v = fmaxf(v, __shfl_xor(v, 1));
  v = fmaxf(v, __shfl_xor(v, 2));
  v = fmaxf(v, __shfl_xor(v, 4));
  v = fmaxf(v, __shfl_xor(v, 8));
  return v;
}
__device__ __forceinline__ float redsum16(float v) {
  v += __shfl_xor(v, 1);
  v += __shfl_xor(v, 2);
  v += __shfl_xor(v, 4);
  v += __shfl_xor(v, 8);
  return v;
}

// Issue one 2D TDM descriptor: copy a [rows x 512] bf16 tile (row-major,
// stride 512) from global to LDS, inserting 8 dwords (32B) of padding after
// every 256 dwords (512 bf16 = one row) -> LDS row stride 528 elements.
__device__ __forceinline__ void tdm_load_tile_bf16(unsigned lds_addr,
                                                   const bf16_t* gptr,
                                                   unsigned rows) {
  unsigned long long ga = (unsigned long long)(uintptr_t)gptr;
  u32x4 g0;
  g0[0] = 1u;                                   // count=1 valid descriptor
  g0[1] = lds_addr;                             // lds_addr [63:32]
  g0[2] = (unsigned)ga;                         // global_addr low
  g0[3] = (unsigned)((ga >> 32) & 0x01FFFFFFu)  // global_addr [56:32]
          | (2u << 30);                         // type=2 ("image")
  i32x8 g1;
  g1[0] = (int)((1u << 16)       // data_size = 1 -> 2 bytes
                | (1u << 20)     // pad_enable
                | (7u << 22)     // pad_interval code 7 = 256 dwords
                | (7u << 25));   // pad_amount  code 7 = 8 dwords (32B)
  g1[1] = (int)(512u << 16);     // tensor_dim0[15:0] << 16 (atomic addr = 0)
  g1[2] = (int)(rows << 16);     // tensor_dim0 hi = 0 | tensor_dim1 lo
  g1[3] = (int)(512u << 16);     // tensor_dim1 hi = 0 | tile_dim0 = 512
  g1[4] = (int)rows;             // tile_dim1 = rows, tile_dim2 = 0
  g1[5] = 512;                   // tensor_dim0_stride low32
  g1[6] = 0;
  g1[7] = 0;
  i32x4 z4;
  z4[0] = 0; z4[1] = 0; z4[2] = 0; z4[3] = 0;   // 2D: groups 2/3 unused
  i32x8 z8;
  z8[0] = 0; z8[1] = 0; z8[2] = 0; z8[3] = 0;
  z8[4] = 0; z8[5] = 0; z8[6] = 0; z8[7] = 0;
  __builtin_amdgcn_tensor_load_to_lds(g0, g1, z4, z4, z8, 0);
}

// ---- kernel 0: fp32 -> bf16 ------------------------------------------------

__global__ void cvt_bf16_kernel(const float* __restrict__ x,
                                bf16_t* __restrict__ y) {
  int i = (blockIdx.x * blockDim.x + threadIdx.x) * 4;
  float4 v = *(const float4*)(x + i);
  bf16x4 o;
  o[0] = (bf16_t)v.x; o[1] = (bf16_t)v.y; o[2] = (bf16_t)v.z; o[3] = (bf16_t)v.w;
  *(bf16x4*)(y + i) = o;
}

// ---- kernel 1: flash attention + residual + LayerNorm1 ---------------------
// grid: (B*C) * (N/64) blocks of 128 threads (4 waves).  Each wave owns 16
// query rows with a full 16x512 fp32 accumulator in VGPRs.  The 32-key K tile
// is DMA'd into LDS by the TDM; V^T is built with a transposing LDS store.

__global__ __launch_bounds__(128, 1)
void attn_ln1_kernel(const float* __restrict__ src,
                     const float* __restrict__ kpm,
                     const float* __restrict__ g1,
                     const float* __restrict__ be1,
                     const bf16_t* __restrict__ xb,
                     bf16_t* __restrict__ s2b) {
  constexpr int LDK = 528;   // K tile stride: 512 + TDM-inserted 32B pad
  constexpr int LDV = 48;    // V^T tile stride (pad, 32B-mult)
  extern __shared__ char smem[];
  bf16_t* kls = (bf16_t*)smem;                                 // [32][LDK]
  bf16_t* vt  = (bf16_t*)(smem + 32 * LDK * sizeof(bf16_t));   // [Ed][LDV]
  bf16_t* pls = (bf16_t*)(smem + (32 * LDK + Ed * LDV) * sizeof(bf16_t));

  const int tid  = threadIdx.x;
  const int lane = tid & 31;
  const int wave = tid >> 5;
  const int ln   = lane & 15;
  const int hi   = lane >> 4;
  const int klo  = hi * 16;   // B-operand k offset
  const int koff = hi * 8;    // unused (A operands also use klo)

  const int blk  = blockIdx.x;
  const int bc   = blk >> 5;           // N/64 = 32 row-blocks per (b,c)
  const int b    = bc / Cd;
  const int q0   = (blk & 31) * 64 + wave * 16;

  const size_t bcoff = (size_t)bc * Nd * Ed;
  const bf16_t* xbc = xb + bcoff;

  // Q block for this wave, pre-packed as 16 A-operands over e-chunks of 32.
  bf16x16 qa[16];
  {
    const bf16_t* qrow = xbc + (size_t)(q0 + ln) * Ed;
    #pragma unroll
    for (int kk = 0; kk < 16; ++kk)
      qa[kk] = ldA(qrow + kk * 32 + klo, qrow + kk * 32 + klo + 8);
  }

  v8f acc[32];                                   // O accumulator 16x512 fp32
  #pragma unroll
  for (int t = 0; t < 32; ++t) {
    #pragma unroll
    for (int r = 0; r < 8; ++r) acc[t][r] = 0.0f;
  }
  float m_run[8], l_run[8];
  #pragma unroll
  for (int r = 0; r < 8; ++r) { m_run[r] = -3.0e38f; l_run[r] = 0.0f; }

  const float scale = 0.044194173824159216f;     // 1/sqrt(512)

  for (int kt = 0; kt < Nd / 32; ++kt) {
    const int key0 = kt * 32;
    __syncthreads();   // previous tile fully consumed

    // Tensor Data Mover: DMA the 32x512 bf16 K tile into LDS with padded
    // rows; single issue from wave 0 (EXEC is ignored by TDM).
    if (wave == 0) {
      tdm_load_tile_bf16((unsigned)(uintptr_t)kls,
                         xbc + (size_t)key0 * Ed, 32u);
    }

    // Meanwhile all waves build V^T: vt[e][m] = xbc[(key0+m)*Ed + e]
    // (coalesced global reads, transposing scalar LDS writes).
    for (int v = tid; v < (32 * Ed) / 8; v += blockDim.x) {
      int m  = v >> 6;            // 64 vec8 per key row
      int e8 = (v & 63) * 8;
      bf16x8 d = *(const bf16x8*)(xbc + (size_t)(key0 + m) * Ed + e8);
      #pragma unroll
      for (int j = 0; j < 8; ++j) vt[(e8 + j) * LDV + m] = d[j];
    }
    if (kt + 1 < Nd / 32) {       // pull next K/V tile toward the caches
      const bf16_t* nxt = xbc + (size_t)(key0 + 32) * Ed;
      __builtin_prefetch(nxt + tid * 128, 0, 0);
    }
    if (wave == 0) __builtin_amdgcn_s_wait_tensorcnt(0);
    __syncthreads();

    // Scores S(16x32) = Q x K^T.  B operand from the TDM-staged LDS tile
    // (contiguous over e, row stride 528 elements -> conflict-free).
    v8f s0, s1;
    #pragma unroll
    for (int r = 0; r < 8; ++r) { s0[r] = 0.0f; s1[r] = 0.0f; }
    {
      const bf16_t* krow0 = kls + ln * LDK + klo;
      const bf16_t* krow1 = kls + (16 + ln) * LDK + klo;
      #pragma unroll
      for (int kk = 0; kk < 16; ++kk) {
        bf16x16 kb0 = *(const bf16x16*)(krow0 + kk * 32);
        bf16x16 kb1 = *(const bf16x16*)(krow1 + kk * 32);
        s0 = wmma_bf16(qa[kk], kb0, s0);
        s1 = wmma_bf16(qa[kk], kb1, s1);
      }
    }

    // Online softmax (rows split: VGPR r, lane-half hi -> row r + 8*hi).
    const float mask0 = kpm[b * Nd + key0 + ln] * MASK_SCALE;
    const float mask1 = kpm[b * Nd + key0 + 16 + ln] * MASK_SCALE;
    float p0[8], p1[8], alpha[8];
    #pragma unroll
    for (int r = 0; r < 8; ++r) {
      float a0 = s0[r] * scale - mask0;
      float a1 = s1[r] * scale - mask1;
      float mx = redmax16(fmaxf(a0, a1));
      float mn = fmaxf(m_run[r], mx);
      alpha[r] = __expf(m_run[r] - mn);
      m_run[r] = mn;
      p0[r] = __expf(a0 - mn);
      p1[r] = __expf(a1 - mn);
      l_run[r] = l_run[r] * alpha[r] + redsum16(p0[r] + p1[r]);
    }
    #pragma unroll
    for (int t = 0; t < 32; ++t) {
      #pragma unroll
      for (int r = 0; r < 8; ++r) acc[t][r] *= alpha[r];
    }

    // P: C-layout -> A-layout via per-wave LDS (DS ops are in-order per wave).
    bf16_t* pw = pls + wave * (16 * 32);
    #pragma unroll
    for (int r = 0; r < 8; ++r) {
      int row = r + 8 * hi;
      pw[row * 32 + ln]      = (bf16_t)p0[r];
      pw[row * 32 + 16 + ln] = (bf16_t)p1[r];
    }
    bf16x16 pa = ldA(pw + ln * 32 + klo, pw + ln * 32 + klo + 8);

    // O += P x V, B operand contiguous from the transposed LDS tile.
    #pragma unroll
    for (int t = 0; t < 32; ++t) {
      bf16x16 bv = *(const bf16x16*)(vt + (t * 16 + ln) * LDV + klo);
      acc[t] = wmma_bf16(pa, bv, acc[t]);
    }
  }

  // Epilogue: normalize, residual, LayerNorm1 -> bf16 src2.
  float inv_l[8];
  #pragma unroll
  for (int r = 0; r < 8; ++r) inv_l[r] = 1.0f / l_run[r];

  const float* srow = src + bcoff;
  float sm1[8], sm2[8];
  #pragma unroll
  for (int r = 0; r < 8; ++r) { sm1[r] = 0.0f; sm2[r] = 0.0f; }
  #pragma unroll
  for (int t = 0; t < 32; ++t) {
    int e = t * 16 + ln;
    #pragma unroll
    for (int r = 0; r < 8; ++r) {
      int row = q0 + r + 8 * hi;
      float x = acc[t][r] * inv_l[r] + srow[(size_t)row * Ed + e];
      acc[t][r] = x;
      sm1[r] += x;
      sm2[r] += x * x;
    }
  }
  #pragma unroll
  for (int r = 0; r < 8; ++r) {
    float mu = redsum16(sm1[r]) * (1.0f / (float)Ed);
    float ms = redsum16(sm2[r]) * (1.0f / (float)Ed);
    sm1[r] = mu;
    sm2[r] = rsqrtf(ms - mu * mu + EPSv);
  }
  bf16_t* orow = s2b + bcoff;
  #pragma unroll
  for (int t = 0; t < 32; ++t) {
    int e = t * 16 + ln;
    float g = g1[e], bb = be1[e];
    #pragma unroll
    for (int r = 0; r < 8; ++r) {
      int row = q0 + r + 8 * hi;
      orow[(size_t)row * Ed + e] =
          (bf16_t)((acc[t][r] - sm1[r]) * sm2[r] * g + bb);
    }
  }
}

// ---- kernel 2: fused FFN (Linear-ReLU-Linear) + residual + LayerNorm2 ------
// Hidden activations never touch HBM: per 32-wide F-chunk, gemm1 -> ReLU ->
// gemm2-accumulate, weights staged fp32->bf16 into LDS.

__global__ __launch_bounds__(128, 1)
void ffn_ln2_kernel(const bf16_t* __restrict__ s2b,
                    const float* __restrict__ W1,
                    const float* __restrict__ b1,
                    const float* __restrict__ W2,
                    const float* __restrict__ b2,
                    const float* __restrict__ g2,
                    const float* __restrict__ be2,
                    float* __restrict__ out) {
  constexpr int LD1 = 528;                      // W1^T chunk stride (32B-mult)
  extern __shared__ char smem[];
  bf16_t* w1c = (bf16_t*)smem;                                  // [32][LD1]
  bf16_t* w2c = (bf16_t*)(smem + 32 * LD1 * sizeof(bf16_t));    // [Ed][32]
  bf16_t* hls = (bf16_t*)(smem + (32 * LD1 + Ed * 32) * sizeof(bf16_t)); // [4][16*32]

  const int tid  = threadIdx.x;
  const int lane = tid & 31;
  const int wave = tid >> 5;
  const int ln   = lane & 15;
  const int hi   = lane >> 4;
  const int klo  = hi * 16;
  const int koff = hi * 8;

  const int blk = blockIdx.x;
  const int bc  = blk >> 5;
  const int q0  = (blk & 31) * 64 + wave * 16;
  const size_t bcoff = (size_t)bc * Nd * Ed;

  // src2 rows for this wave as 16 pre-packed A-operands over e.
  bf16x16 sa[16];
  {
    const bf16_t* srow2 = s2b + bcoff + (size_t)(q0 + ln) * Ed;
    #pragma unroll
    for (int kk = 0; kk < 16; ++kk)
      sa[kk] = ldA(srow2 + kk * 32 + klo, srow2 + kk * 32 + klo + 8);
  }

  v8f acc[32];                                   // out accumulator 16x512 fp32
  #pragma unroll
  for (int t = 0; t < 32; ++t) {
    #pragma unroll
    for (int r = 0; r < 8; ++r) acc[t][r] = 0.0f;
  }

  for (int fc = 0; fc < Fd / 32; ++fc) {
    const int f0 = fc * 32;
    __syncthreads();
    // Stage W1^T chunk: w1c[f][e] = W1[e, f0+f]  (fp32 -> bf16 on the fly).
    for (int idx = tid; idx < 32 * Ed; idx += blockDim.x) {
      int e = idx >> 5, f = idx & 31;
      w1c[f * LD1 + e] = (bf16_t)W1[(size_t)e * Fd + f0 + f];
    }
    // Stage W2^T chunk: w2c[e][f] = W2[f0+f, e].
    for (int idx = tid; idx < 32 * Ed; idx += blockDim.x) {
      int f = idx >> 9, e = idx & 511;
      w2c[e * 32 + f] = (bf16_t)W2[(size_t)(f0 + f) * Ed + e];
    }
    if (fc + 1 < Fd / 32)
      __builtin_prefetch(W2 + (size_t)(f0 + 32) * Ed + tid * 128, 0, 0);
    __syncthreads();

    // gemm1: h(16x32) = src2(16x512) x W1[:, f-chunk]
    v8f h0, h1;
    #pragma unroll
    for (int r = 0; r < 8; ++r) { h0[r] = 0.0f; h1[r] = 0.0f; }
    #pragma unroll
    for (int kk = 0; kk < 16; ++kk) {
      bf16x16 wb0 = *(const bf16x16*)(w1c + ln * LD1 + kk * 32 + klo);
      bf16x16 wb1 = *(const bf16x16*)(w1c + (16 + ln) * LD1 + kk * 32 + klo);
      h0 = wmma_bf16(sa[kk], wb0, h0);
      h1 = wmma_bf16(sa[kk], wb1, h1);
    }

    // bias + ReLU, C-layout -> A-layout via per-wave LDS.
    const float bb0 = b1[f0 + ln];
    const float bb1 = b1[f0 + 16 + ln];
    bf16_t* hw = hls + wave * (16 * 32);
    #pragma unroll
    for (int r = 0; r < 8; ++r) {
      int row = r + 8 * hi;
      hw[row * 32 + ln]      = (bf16_t)fmaxf(h0[r] + bb0, 0.0f);
      hw[row * 32 + 16 + ln] = (bf16_t)fmaxf(h1[r] + bb1, 0.0f);
    }
    bf16x16 ha = ldA(hw + ln * 32 + klo, hw + ln * 32 + klo + 8);

    // gemm2: acc(16x512) += h(16x32) x W2[f-chunk, :]
    #pragma unroll
    for (int t = 0; t < 32; ++t) {
      bf16x16 bw = *(const bf16x16*)(w2c + (t * 16 + ln) * 32 + klo);
      acc[t] = wmma_bf16(ha, bw, acc[t]);
    }
  }

  // Epilogue: + b2 + residual(src2), LayerNorm2, fp32 store.
  float sm1[8], sm2[8];
  #pragma unroll
  for (int r = 0; r < 8; ++r) { sm1[r] = 0.0f; sm2[r] = 0.0f; }
  #pragma unroll
  for (int t = 0; t < 32; ++t) {
    int e = t * 16 + ln;
    float bb = b2[e];
    #pragma unroll
    for (int r = 0; r < 8; ++r) {
      int row = q0 + r + 8 * hi;
      float x = acc[t][r] + bb + (float)s2b[bcoff + (size_t)row * Ed + e];
      acc[t][r] = x;
      sm1[r] += x;
      sm2[r] += x * x;
    }
  }
  #pragma unroll
  for (int r = 0; r < 8; ++r) {
    float mu = redsum16(sm1[r]) * (1.0f / (float)Ed);
    float ms = redsum16(sm2[r]) * (1.0f / (float)Ed);
    sm1[r] = mu;
    sm2[r] = rsqrtf(ms - mu * mu + EPSv);
  }
  float* orow = out + bcoff;
  #pragma unroll
  for (int t = 0; t < 32; ++t) {
    int e = t * 16 + ln;
    float g = g2[e], bb = be2[e];
    #pragma unroll
    for (int r = 0; r < 8; ++r) {
      int row = q0 + r + 8 * hi;
      orow[(size_t)row * Ed + e] = (acc[t][r] - sm1[r]) * sm2[r] * g + bb;
    }
  }
}

// ---- launch ----------------------------------------------------------------

extern "C" void kernel_launch(void* const* d_in, const int* in_sizes, int n_in,
                              void* d_out, int out_size, void* d_ws,
                              size_t ws_size, hipStream_t stream) {
  const float* src = (const float*)d_in[0];
  const float* kpm = (const float*)d_in[1];
  const float* W1  = (const float*)d_in[2];
  const float* b1  = (const float*)d_in[3];
  const float* W2  = (const float*)d_in[4];
  const float* b2  = (const float*)d_in[5];
  const float* g1  = (const float*)d_in[6];
  const float* be1 = (const float*)d_in[7];
  const float* g2  = (const float*)d_in[8];
  const float* be2 = (const float*)d_in[9];
  float* out = (float*)d_out;

  const size_t total = (size_t)Bd * Cd * Nd * Ed;   // 33,554,432
  bf16_t* xb  = (bf16_t*)d_ws;                      // 64 MB
  bf16_t* s2b = xb + total;                         // 64 MB

  // 0) src fp32 -> bf16
  cvt_bf16_kernel<<<(int)(total / 4 / 256), 256, 0, stream>>>(src, xb);

  // 1) flash attention + residual + LN1  (1024 WGs of 4 waves)
  const int nblk = Bd * Cd * (Nd / 64);             // 1024
  size_t shm1 = (size_t)(32 * 528 + Ed * 48 + 4 * 16 * 32) * sizeof(bf16_t);
  attn_ln1_kernel<<<nblk, 128, shm1, stream>>>(src, kpm, g1, be1, xb, s2b);

  // 2) fused FFN + residual + LN2
  size_t shm2 = (size_t)(32 * 528 + Ed * 32 + 4 * 16 * 32) * sizeof(bf16_t);
  ffn_ln2_kernel<<<nblk, 128, shm2, stream>>>(s2b, W1, b1, W2, b2, g2, be2, out);
}
